// GlobalForceNet_37082747634272
// MI455X (gfx1250) — matrix operand order repaired and checked by reading, not compiled, not run on hardware
//
#include <hip/hip_runtime.h>
#include <hip/hip_bf16.h>

// Problem constants (match reference)
#define NN      8192
#define INDIM   256
#define HID     512
#define CD      128      // HID/4
#define COMB    640      // HID + CD
#define TOPK    32

// GEMM tiling
#define BM      128
#define BN      128
#define KT      32
#define AS_STRIDE 40     // halfs per As row (32 + 8 pad) = 20 dwords
#define BT_STRIDE 40     // halfs per BsT column (32 + 8 pad) = 20 dwords

typedef _Float16 v16h __attribute__((ext_vector_type(16)));
typedef _Float16 v8h  __attribute__((ext_vector_type(8)));
typedef _Float16 v2h  __attribute__((ext_vector_type(2)));
typedef float    v8f  __attribute__((ext_vector_type(8)));

// ---------------------------------------------------------------------------
// C[M,N] = A[M,Kdim] @ W[Kdim,N] + bias, fp32 in/out, f16 WMMA inner product.
// Block: 256 threads (8 waves). Block tile: 128x128. Wave tile: 16x128.
// LDS layouts make every WMMA fragment two contiguous ds_load_b128 per lane:
//   As  row-major  [row][k]   -> A frag = row's halfs [hi*8..+8) and [16+hi*8..+8)
//   BsT col-major  [col][k]   -> B frag = column's halfs [hi*8..+8) and [16+hi*8..+8)
// All B fragments are preloaded before the WMMA chain so the 8 WMMAs issue
// back-to-back behind one s_wait_dscnt instead of one drain per WMMA.
// ---------------------------------------------------------------------------
__global__ __launch_bounds__(256)
void gfn_gemm_wmma(const float* __restrict__ A, int lda,
                   const float* __restrict__ W, int ldw,
                   const float* __restrict__ bias,
                   float* __restrict__ C, int ldc, int Kdim) {
    const int tid  = threadIdx.x;
    const int lane = tid & 31;
    const int wv   = tid >> 5;
    const int m    = lane & 15;     // row/col within 16
    const int hi   = lane >> 4;     // half-wave select

    const int rowBase = blockIdx.y * BM;
    const int colBase = blockIdx.x * BN;

    __shared__ __align__(16) _Float16 As[BM * AS_STRIDE];
    __shared__ __align__(16) _Float16 BsT[BN * BT_STRIDE];

    v8f acc[8] = {};

    // Per-thread staging coordinates (compile-time trip counts below)
    const int ak2 = tid & 15;             // k-pair 0..15
    const int ar0 = tid >> 4;             // starting row 0..15 (stride 16)
    const int bc0 = tid & 127;            // column 0..127
    const int bkg = tid >> 7;             // k-pair group 0..1

    for (int kp = 0; kp < Kdim; kp += KT) {
        // --- Stage A panel (128 rows x 32 k): 8 unrolled float2 loads ---
        {
            const float* ap = A + (size_t)(rowBase + ar0) * lda + (kp + 2 * ak2);
            _Float16*    as = As + ar0 * AS_STRIDE + 2 * ak2;
            float2 av[8];
#pragma unroll
            for (int it = 0; it < 8; ++it)
                av[it] = *(const float2*)(ap + (size_t)(it * 16) * lda);
#pragma unroll
            for (int it = 0; it < 8; ++it) {
                v2h p; p.x = (_Float16)av[it].x; p.y = (_Float16)av[it].y;
                *(v2h*)(as + it * 16 * AS_STRIDE) = p;
            }
        }
        // --- Stage W panel transposed (col-major): 16 unrolled loads ---
        {
            const float* wp = W + (size_t)(kp + 2 * bkg) * ldw + (colBase + bc0);
            _Float16*    bs = BsT + bc0 * BT_STRIDE + 2 * bkg;
            float w0[8], w1[8];
#pragma unroll
            for (int it = 0; it < 8; ++it) {
                w0[it] = wp[(size_t)(it * 4 + 0) * ldw];
                w1[it] = wp[(size_t)(it * 4 + 1) * ldw];
            }
#pragma unroll
            for (int it = 0; it < 8; ++it) {
                v2h p; p.x = (_Float16)w0[it]; p.y = (_Float16)w1[it];
                *(v2h*)(bs + it * 4) = p;
            }
        }
        // Prefetch next A panel into cache hierarchy (global_prefetch_b8)
        if (kp + KT < Kdim)
            __builtin_prefetch(&A[(size_t)(rowBase + (tid & (BM - 1))) * lda + kp + KT], 0, 1);
        __syncthreads();

        // A fragment: two contiguous 8-half runs -> 2x ds_load_b128
        const _Float16* ar = &As[(wv * 16 + m) * AS_STRIDE];
        v8h alo = *(const v8h*)(ar + hi * 8);
        v8h ahi = *(const v8h*)(ar + 16 + hi * 8);
        v16h a = __builtin_shufflevector(alo, ahi,
                 0, 1, 2, 3, 4, 5, 6, 7, 8, 9, 10, 11, 12, 13, 14, 15);

        // Preload ALL B fragments (16x ds_load_b128), then run the WMMA chain
        v16h bfrag[8];
#pragma unroll
        for (int t = 0; t < 8; ++t) {
            const _Float16* bc = &BsT[(t * 16 + m) * BT_STRIDE + hi * 8];
            v8h blo = *(const v8h*)(bc);
            v8h bhi = *(const v8h*)(bc + 16);
            bfrag[t] = __builtin_shufflevector(blo, bhi,
                       0, 1, 2, 3, 4, 5, 6, 7, 8, 9, 10, 11, 12, 13, 14, 15);
        }
#pragma unroll
        for (int t = 0; t < 8; ++t)
            acc[t] = __builtin_amdgcn_wmma_f32_16x16x32_f16(
                false, a, false, bfrag[t], (short)0, acc[t], false, false);
        __syncthreads();
    }

    // Epilogue: VGPR r of C tile -> M = r (lanes 0-15) / r+8 (lanes 16-31)
    const int rowTop = rowBase + wv * 16 + hi * 8;
#pragma unroll
    for (int t = 0; t < 8; ++t) {
        int col = colBase + t * 16 + m;
        float bv = bias[col];
#pragma unroll
        for (int r = 0; r < 8; ++r)
            C[(size_t)(rowTop + r) * ldc + col] = acc[t][r] + bv;
    }
}

// ---------------------------------------------------------------------------
// h[:, 512:640] = coords @ W_coord + b_coord   (K=2, pure VALU)
// ---------------------------------------------------------------------------
__global__ __launch_bounds__(256)
void gfn_coord_embed(const float* __restrict__ coords,
                     const float* __restrict__ Wc,
                     const float* __restrict__ bc,
                     float* __restrict__ h) {
    int t = blockIdx.x * blockDim.x + threadIdx.x;
    if (t >= NN * CD) return;
    int i = t >> 7;          // node
    int j = t & (CD - 1);    // coord-feature
    float c0 = coords[i * 2 + 0], c1 = coords[i * 2 + 1];
    h[(size_t)i * COMB + HID + j] = c0 * Wc[j] + c1 * Wc[CD + j] + bc[j];
}

// ---------------------------------------------------------------------------
// Per-row top-32 smallest 2D distances. One workgroup per row.
// d2 for all 8192 candidates lives in LDS; 32x argmin with cached local mins.
// ---------------------------------------------------------------------------
__global__ __launch_bounds__(256)
void gfn_topk(const float* __restrict__ coords,
              int* __restrict__ nbr, float* __restrict__ dtk) {
    const int row = blockIdx.x;
    const int tid = threadIdx.x;

    __shared__ float d2buf[NN];          // 32 KB
    __shared__ float rv[256];
    __shared__ int   ri[256];
    __shared__ float lmin[256];
    __shared__ int   lidx[256];

    const float cx = coords[row * 2 + 0];
    const float cy = coords[row * 2 + 1];

    float bestv = 3.0e38f;
    int   bestj = tid;
#pragma unroll 4
    for (int i = 0; i < NN / 256; ++i) {
        int j = i * 256 + tid;           // bank-conflict-free column layout
        float dx = coords[j * 2 + 0] - cx;
        float dy = coords[j * 2 + 1] - cy;
        float d2 = dx * dx + dy * dy;
        d2buf[j] = d2;
        if (d2 < bestv) { bestv = d2; bestj = j; }
    }
    lmin[tid] = bestv; lidx[tid] = bestj;
    __syncthreads();

    for (int s = 0; s < TOPK; ++s) {
        rv[tid] = lmin[tid]; ri[tid] = lidx[tid];
        __syncthreads();
        for (int off = 128; off > 0; off >>= 1) {
            if (tid < off && rv[tid + off] < rv[tid]) {
                rv[tid] = rv[tid + off]; ri[tid] = ri[tid + off];
            }
            __syncthreads();
        }
        const int   jwin = ri[0];
        const float vwin = rv[0];
        if (tid == 0) {
            nbr[row * TOPK + s] = jwin;
            float d2 = fmaxf(vwin, 0.0f);
            dtk[row * TOPK + s] = (d2 > 0.0f) ? sqrtf(d2) : 0.0f;
        }
        if ((jwin & 255) == tid) {       // owner removes winner, rescans its 32
            d2buf[jwin] = 3.0e38f;
            float bv = 3.0e38f; int bj = tid;
#pragma unroll 4
            for (int i = 0; i < NN / 256; ++i) {
                int j = i * 256 + tid;
                float v = d2buf[j];
                if (v < bv) { bv = v; bj = j; }
            }
            lmin[tid] = bv; lidx[tid] = bj;
        }
        __syncthreads();
    }
}

// ---------------------------------------------------------------------------
// Attention over 32 gathered neighbors. One workgroup per row.
// ---------------------------------------------------------------------------
__global__ __launch_bounds__(256)
void gfn_attn(const float* __restrict__ Q, const float* __restrict__ Km,
              const float* __restrict__ V, const int* __restrict__ nbr,
              const float* __restrict__ dtk, float* __restrict__ g) {
    const int row = blockIdx.x;
    const int tid = threadIdx.x;

    __shared__ __align__(16) float qs[HID];
    __shared__ float sc[TOPK];
    __shared__ float wts[TOPK];
    __shared__ int   nb[TOPK];

    qs[tid]       = Q[(size_t)row * HID + tid];
    qs[tid + 256] = Q[(size_t)row * HID + tid + 256];
    if (tid < TOPK) nb[tid] = nbr[row * TOPK + tid];
    __syncthreads();

    const int s    = tid >> 3;   // neighbor slot 0..31
    const int part = tid & 7;    // 8 threads per slot, group within one wave
    {
        const float* krow = Km + (size_t)nb[s] * HID;
        float sum = 0.0f;
        const float4* k4 = (const float4*)krow;
        const float4* q4 = (const float4*)qs;
#pragma unroll
        for (int e4 = part * 16; e4 < part * 16 + 16; ++e4) {
            float4 kv = k4[e4], qv = q4[e4];
            sum += kv.x * qv.x + kv.y * qv.y + kv.z * qv.z + kv.w * qv.w;
        }
        sum += __shfl_down(sum, 4, 8);
        sum += __shfl_down(sum, 2, 8);
        sum += __shfl_down(sum, 1, 8);
        if (part == 0)
            sc[s] = sum * 0.04419417382415922f        // 1/sqrt(512)
                    + 1.0f / (dtk[row * TOPK + s] + 1e-6f);
    }
    __syncthreads();

    if (tid < TOPK) {                                 // wave softmax over 32
        float v = sc[tid];
        float mx = v;
        for (int off = 16; off >= 1; off >>= 1) mx = fmaxf(mx, __shfl_xor(mx, off, 32));
        float e = __expf(v - mx);
        float sm = e;
        for (int off = 16; off >= 1; off >>= 1) sm += __shfl_xor(sm, off, 32);
        wts[tid] = e / sm;
    }
    __syncthreads();

    for (int e = tid; e < HID; e += 256) {
        float acc = 0.0f;
#pragma unroll
        for (int s2 = 0; s2 < TOPK; ++s2)
            acc += wts[s2] * V[(size_t)nb[s2] * HID + e];   // coalesced across e
        g[(size_t)row * HID + e] = acc;
    }
}

// ---------------------------------------------------------------------------
// out[N,2] = g @ W_f + b_f. One wave per row, shuffle reduction.
// ---------------------------------------------------------------------------
__global__ __launch_bounds__(256)
void gfn_final(const float* __restrict__ g, const float* __restrict__ Wf,
               const float* __restrict__ bf, float* __restrict__ out) {
    const int lane = threadIdx.x & 31;
    const int row  = (blockIdx.x * blockDim.x + threadIdx.x) >> 5;
    if (row >= NN) return;
    float s0 = 0.0f, s1 = 0.0f;
    for (int e = lane; e < HID; e += 32) {
        float gv = g[(size_t)row * HID + e];
        s0 += gv * Wf[e * 2 + 0];
        s1 += gv * Wf[e * 2 + 1];
    }
    for (int off = 16; off >= 1; off >>= 1) {
        s0 += __shfl_xor(s0, off, 32);
        s1 += __shfl_xor(s1, off, 32);
    }
    if (lane == 0) {
        out[row * 2 + 0] = s0 + bf[0];
        out[row * 2 + 1] = s1 + bf[1];
    }
}

// ---------------------------------------------------------------------------
extern "C" void kernel_launch(void* const* d_in, const int* in_sizes, int n_in,
                              void* d_out, int out_size, void* d_ws, size_t ws_size,
                              hipStream_t stream) {
    const float* x       = (const float*)d_in[0];
    // d_in[1] = edge_index (int64) — unused by reference forward
    const float* coords  = (const float*)d_in[2];
    const float* W_node  = (const float*)d_in[3];
    const float* b_node  = (const float*)d_in[4];
    const float* W_coord = (const float*)d_in[5];
    const float* b_coord = (const float*)d_in[6];
    const float* W_q     = (const float*)d_in[7];
    const float* b_q     = (const float*)d_in[8];
    const float* W_k     = (const float*)d_in[9];
    const float* b_k     = (const float*)d_in[10];
    const float* W_v     = (const float*)d_in[11];
    const float* b_v     = (const float*)d_in[12];
    const float* W_f     = (const float*)d_in[13];
    const float* b_f     = (const float*)d_in[14];
    float* out = (float*)d_out;

    // Workspace carve-up (~86 MB)
    float* h   = (float*)d_ws;                       // [N, 640]
    float* Q   = h  + (size_t)NN * COMB;             // [N, 512]
    float* Kt  = Q  + (size_t)NN * HID;
    float* V   = Kt + (size_t)NN * HID;
    float* g   = V  + (size_t)NN * HID;
    int*   nbr = (int*)(g + (size_t)NN * HID);       // [N, 32]
    float* dtk = (float*)(nbr + (size_t)NN * TOPK);  // [N, 32]

    dim3 blk(256);
    dim3 gemmGrid(HID / BN, NN / BM);                // (4, 64)

    // h_node = x @ W_node + b_node  -> h[:, 0:512]
    gfn_gemm_wmma<<<gemmGrid, blk, 0, stream>>>(x, INDIM, W_node, HID, b_node,
                                                h, COMB, INDIM);
    // h_coord -> h[:, 512:640]
    gfn_coord_embed<<<(NN * CD) / 256, blk, 0, stream>>>(coords, W_coord, b_coord, h);

    // Q/K/V = h @ W_{q,k,v} + b
    gfn_gemm_wmma<<<gemmGrid, blk, 0, stream>>>(h, COMB, W_q, HID, b_q, Q,  HID, COMB);
    gfn_gemm_wmma<<<gemmGrid, blk, 0, stream>>>(h, COMB, W_k, HID, b_k, Kt, HID, COMB);
    gfn_gemm_wmma<<<gemmGrid, blk, 0, stream>>>(h, COMB, W_v, HID, b_v, V,  HID, COMB);

    // top-32 nearest neighbors per row (2D coords)
    gfn_topk<<<NN, blk, 0, stream>>>(coords, nbr, dtk);

    // gathered attention
    gfn_attn<<<NN, blk, 0, stream>>>(Q, Kt, V, nbr, dtk, g);

    // final projection to [N, 2]
    gfn_final<<<NN / 8, blk, 0, stream>>>(g, W_f, b_f, out);
}
